// AttentionBlock_68006512165454
// MI455X (gfx1250) — matrix operand order repaired and checked by reading, not compiled, not run on hardware
//
#include <hip/hip_runtime.h>
#include <hip/hip_bf16.h>

#define Bdim 8
#define Tdim 2048
#define Cdim 2048
#define Mrows (Bdim * Tdim)
#define LN_EPS 1e-5f

typedef __attribute__((ext_vector_type(16))) __bf16 v16bf;
typedef __attribute__((ext_vector_type(8)))  float  v8f;

static __device__ __forceinline__ __bf16 f2bf(float f) {
    union { float f; unsigned u; } a; a.f = f;
    unsigned r = a.u + 0x7FFFu + ((a.u >> 16) & 1u);   // round-to-nearest-even
    unsigned short h = (unsigned short)(r >> 16);
    union { unsigned short s; __bf16 b; } o; o.s = h;
    return o.b;
}

// CDNA5 async copy: global -> LDS, 16 bytes per lane, tracked by ASYNCcnt.
static __device__ __forceinline__ void async_ld_b128(unsigned lds_off, const void* g) {
    asm volatile("global_load_async_to_lds_b128 %0, %1, off"
                 :: "v"(lds_off), "v"(g) : "memory");
}
static __device__ __forceinline__ void wait_async0() {
    asm volatile("s_wait_asynccnt 0" ::: "memory");
}
// low 32 bits of a generic pointer into LDS == LDS byte offset (ISA §10.2 aperture rule)
static __device__ __forceinline__ unsigned lds_off_of(const void* p) {
    return (unsigned)(unsigned long long)p;
}

// ---------------------------------------------------------------------------
// fp32 -> bf16 convert (weights)
// ---------------------------------------------------------------------------
__global__ __launch_bounds__(256) void cvt_f32_bf16(const float* __restrict__ src,
                                                    __bf16* __restrict__ dst, long long n) {
    long long i = (long long)blockIdx.x * blockDim.x + threadIdx.x;
    long long stride = (long long)gridDim.x * blockDim.x;
    for (; i < n; i += stride) dst[i] = f2bf(src[i]);
}

// ---------------------------------------------------------------------------
// LayerNorm row statistics: one block per (b,t) row of C elements
// ---------------------------------------------------------------------------
__global__ __launch_bounds__(256) void ln_stats(const float* __restrict__ x,
                                                float* __restrict__ mu,
                                                float* __restrict__ rstd) {
    int row = blockIdx.x;
    const float* xr = x + (size_t)row * Cdim;
    float s = 0.f, s2 = 0.f;
    for (int c = threadIdx.x; c < Cdim; c += 256) {
        float v = xr[c]; s += v; s2 += v * v;
    }
    __shared__ float sh0[256], sh1[256];
    sh0[threadIdx.x] = s; sh1[threadIdx.x] = s2;
    __syncthreads();
    for (int st = 128; st > 0; st >>= 1) {
        if (threadIdx.x < st) {
            sh0[threadIdx.x] += sh0[threadIdx.x + st];
            sh1[threadIdx.x] += sh1[threadIdx.x + st];
        }
        __syncthreads();
    }
    if (threadIdx.x == 0) {
        float m = sh0[0] * (1.0f / Cdim);
        float var = sh1[0] * (1.0f / Cdim) - m * m;
        mu[row] = m;
        rstd[row] = rsqrtf(var + LN_EPS);
    }
}

// ---------------------------------------------------------------------------
// Fused normalize + token-shift + time-mix -> bf16 xk/xv/xr
// ---------------------------------------------------------------------------
__global__ __launch_bounds__(256) void mix_kernel(
    const float* __restrict__ x, const float* __restrict__ mu, const float* __restrict__ rstd,
    const float* __restrict__ g, const float* __restrict__ beta,
    const float* __restrict__ tmk, const float* __restrict__ tmv, const float* __restrict__ tmr,
    const float* __restrict__ xx0,
    __bf16* __restrict__ xk, __bf16* __restrict__ xv, __bf16* __restrict__ xr) {
    size_t idx = (size_t)blockIdx.x * blockDim.x + threadIdx.x;
    const size_t total = (size_t)Mrows * Cdim;
    if (idx >= total) return;
    int c = (int)(idx % Cdim);
    size_t row = idx / Cdim;
    int t = (int)(row % Tdim);
    int b = (int)(row / Tdim);

    float gc = g[c], bc = beta[c];
    float ln_t = (x[idx] - mu[row]) * rstd[row] * gc + bc;
    float prev;
    if (t == 0) {
        prev = xx0[(size_t)b * Cdim + c];
    } else {
        size_t prow = row - 1;
        prev = (x[idx - Cdim] - mu[prow]) * rstd[prow] * gc + bc;
    }
    float mk = tmk[c], mv = tmv[c], mr = tmr[c];
    xk[idx] = f2bf(ln_t * mk + prev * (1.0f - mk));
    xv[idx] = f2bf(ln_t * mv + prev * (1.0f - mv));
    xr[idx] = f2bf(ln_t * mr + prev * (1.0f - mr));
}

// ---------------------------------------------------------------------------
// bf16 WMMA GEMM with double-buffered async global->LDS staging.
// out[m,n] = epilogue( sum_k A[m,k] * W[n,k] )
// EPI: 0 = plain, 1 = sigmoid, 2 = addend[m,n] + acc
// Block = 256 threads (8 waves), tile 128x128, K-step 32.
// ---------------------------------------------------------------------------
#define BM 128
#define BN 128
#define BK 32
#define LDSTR 40   // 32 + 8 bf16 pad to dodge LDS bank conflicts

template <int EPI>
__global__ __launch_bounds__(256) void gemm_bf16(
    const __bf16* __restrict__ A, const __bf16* __restrict__ W,
    const float* __restrict__ addend, float* __restrict__ out,
    int M, int N, int K) {
    __shared__ __align__(16) __bf16 lA[2][BM * LDSTR];
    __shared__ __align__(16) __bf16 lB[2][BN * LDSTR];

    const int tid  = threadIdx.x;
    const int lane = tid & 31;
    const int wave = tid >> 5;
    const int half = lane >> 4;
    const int l16  = lane & 15;

    const int bm = blockIdx.x * BM;
    const int bn = blockIdx.y * BN;
    const int wm = (wave & 3) * 32;   // 4 waves over M
    const int wn = (wave >> 2) * 64;  // 2 waves over N

    // --- per-thread staging assignment: 2 x b128 chunks per tile ------------
    // chunk i (0..511): row = i>>2, col8 = (i&3)*8 bf16
    const int r0 = tid >> 2,        c0 = (tid & 3) * 8;
    const int r1 = (tid >> 2) + 64; const int c1 = c0;
    const char* gA0 = (const char*)&A[(size_t)(bm + r0) * K + c0];
    const char* gA1 = (const char*)&A[(size_t)(bm + r1) * K + c1];
    const char* gB0 = (const char*)&W[(size_t)(bn + r0) * K + c0];
    const char* gB1 = (const char*)&W[(size_t)(bn + r1) * K + c1];
    unsigned laA0[2], laA1[2], laB0[2], laB1[2];
#pragma unroll
    for (int s = 0; s < 2; ++s) {
        laA0[s] = lds_off_of(&lA[s][r0 * LDSTR + c0]);
        laA1[s] = lds_off_of(&lA[s][r1 * LDSTR + c1]);
        laB0[s] = lds_off_of(&lB[s][r0 * LDSTR + c0]);
        laB1[s] = lds_off_of(&lB[s][r1 * LDSTR + c1]);
    }

    const v8f vzero = {0.f, 0.f, 0.f, 0.f, 0.f, 0.f, 0.f, 0.f};
    v8f acc[2][4];
#pragma unroll
    for (int i = 0; i < 2; ++i)
#pragma unroll
        for (int j = 0; j < 4; ++j) acc[i][j] = vzero;

    const int nk = K / BK;
    // prologue: stage tile 0 into buffer 0
    async_ld_b128(laA0[0], gA0);
    async_ld_b128(laA1[0], gA1);
    async_ld_b128(laB0[0], gB0);
    async_ld_b128(laB1[0], gB1);

    for (int kt = 0; kt < nk; ++kt) {
        wait_async0();          // this wave's staged tile is in LDS
        __syncthreads();        // ... and so is every other wave's

        if (kt + 1 < nk) {      // stage next tile into the other buffer
            int nb = (kt + 1) & 1;
            size_t koff = (size_t)(kt + 1) * BK * 2;   // bytes
            async_ld_b128(laA0[nb], gA0 + koff);
            async_ld_b128(laA1[nb], gA1 + koff);
            async_ld_b128(laB0[nb], gB0 + koff);
            async_ld_b128(laB1[nb], gB1 + koff);
        }

        const __bf16* bufA = lA[kt & 1];
        const __bf16* bufB = lB[kt & 1];

        // A fragments: 16x32 bf16, half 0 holds K 0..15, half 1 K 16..31.
        v16bf afrag[2];
#pragma unroll
        for (int i = 0; i < 2; ++i) {
            int m = wm + i * 16 + l16;
            union { v16bf v; uint4 q[2]; } u;
            u.q[0] = *(const uint4*)&bufA[m * LDSTR + half * 16];
            u.q[1] = *(const uint4*)&bufA[m * LDSTR + half * 16 + 8];
            afrag[i] = u.v;
        }
        // B fragments: 32x16 (KxN), lane n = l16, half selects K 0..15 / 16..31.
        v16bf bfrag[4];
#pragma unroll
        for (int j = 0; j < 4; ++j) {
            int n = wn + j * 16 + l16;
            union { v16bf v; uint4 q[2]; } u;
            u.q[0] = *(const uint4*)&bufB[n * LDSTR + half * 16];
            u.q[1] = *(const uint4*)&bufB[n * LDSTR + half * 16 + 8];
            bfrag[j] = u.v;
        }

#pragma unroll
        for (int i = 0; i < 2; ++i)
#pragma unroll
            for (int j = 0; j < 4; ++j)
                acc[i][j] = __builtin_amdgcn_wmma_f32_16x16x32_bf16(
                    false, afrag[i], false, bfrag[j], (short)0, acc[i][j],
                    false, false);
        // no trailing barrier: next overwrite of buf[kt&1] is issued only after
        // every wave passes the top-of-loop barrier of the next iteration,
        // and this wave's LDS reads completed before it reaches that barrier.
    }

    // Epilogue: C/D layout — VGPR r: m = r + 8*half, n = l16.
#pragma unroll
    for (int i = 0; i < 2; ++i)
#pragma unroll
        for (int j = 0; j < 4; ++j)
#pragma unroll
            for (int r = 0; r < 8; ++r) {
                int m = bm + wm + i * 16 + half * 8 + r;
                int n = bn + wn + j * 16 + l16;
                float v = acc[i][j][r];
                if (EPI == 1) v = 1.0f / (1.0f + __expf(-v));
                if (EPI == 2) v += addend[(size_t)m * N + n];
                out[(size_t)m * N + n] = v;
            }
}

// ---------------------------------------------------------------------------
// WKV sequential scan: one thread per (b,c) channel, loops over T.
// ---------------------------------------------------------------------------
__global__ __launch_bounds__(256) void wkv_scan(
    const float* __restrict__ kk, const float* __restrict__ vv,
    const float* __restrict__ time_first, const float* __restrict__ time_decay,
    const float* __restrict__ aa0, const float* __restrict__ bb0,
    const float* __restrict__ pp0, float* __restrict__ wkv) {
    int idx = blockIdx.x * blockDim.x + threadIdx.x;   // over B*C
    if (idx >= Bdim * Cdim) return;
    int c = idx % Cdim;
    int b = idx / Cdim;
    float aa = aa0[idx], bb = bb0[idx], pp = pp0[idx];
    const float tf = time_first[c], td = time_decay[c];
    size_t base = (size_t)b * Tdim * Cdim + c;
    for (int t = 0; t < Tdim; ++t) {
        size_t off = base + (size_t)t * Cdim;
        float kt = kk[off], vt = vv[off];
        float ww = tf + kt;
        float p  = fmaxf(pp, ww);
        float e1 = __expf(pp - p), e2 = __expf(ww - p);
        wkv[off] = (e1 * aa + e2 * vt) / (e1 * bb + e2);
        float ww2 = pp + td;
        float p2  = fmaxf(ww2, kt);
        float e1b = __expf(ww2 - p2), e2b = __expf(kt - p2);
        aa = e1b * aa + e2b * vt;
        bb = e1b * bb + e2b;
        pp = p2;
    }
}

// ---------------------------------------------------------------------------
// rwkv = bf16( sigmoid(r) * wkv )   (r already sigmoided in GEMM epilogue)
// ---------------------------------------------------------------------------
__global__ __launch_bounds__(256) void rmul_kernel(const float* __restrict__ r,
                                                   const float* __restrict__ wkv,
                                                   __bf16* __restrict__ dst) {
    size_t idx = (size_t)blockIdx.x * blockDim.x + threadIdx.x;
    const size_t total = (size_t)Mrows * Cdim;
    if (idx >= total) return;
    dst[idx] = f2bf(r[idx] * wkv[idx]);
}

// ---------------------------------------------------------------------------
extern "C" void kernel_launch(void* const* d_in, const int* in_sizes, int n_in,
                              void* d_out, int out_size, void* d_ws, size_t ws_size,
                              hipStream_t stream) {
    const float* x    = (const float*)d_in[0];
    const float* g    = (const float*)d_in[1];
    const float* beta = (const float*)d_in[2];
    const float* tmk  = (const float*)d_in[3];
    const float* tmv  = (const float*)d_in[4];
    const float* tmr  = (const float*)d_in[5];
    const float* tf   = (const float*)d_in[6];
    const float* td   = (const float*)d_in[7];
    const float* kw   = (const float*)d_in[8];
    const float* vw   = (const float*)d_in[9];
    const float* rw   = (const float*)d_in[10];
    const float* ow   = (const float*)d_in[11];
    const float* xx0  = (const float*)d_in[12];
    const float* aa0  = (const float*)d_in[13];
    const float* bb0  = (const float*)d_in[14];
    const float* pp0  = (const float*)d_in[15];
    float* out = (float*)d_out;

    const long long MC = (long long)Mrows * Cdim;   // 33.5M
    const long long CC = (long long)Cdim * Cdim;    // 4.2M

    char* ws = (char*)d_ws;
    size_t off = 0;
    auto alloc = [&](size_t bytes) -> void* {
        void* p = ws + off;
        off = (off + bytes + 255) & ~(size_t)255;
        return p;
    };
    float*  mu    = (float*)alloc(Mrows * 4);
    float*  rstd  = (float*)alloc(Mrows * 4);
    __bf16* xk    = (__bf16*)alloc(MC * 2);
    __bf16* xv    = (__bf16*)alloc(MC * 2);
    __bf16* xr    = (__bf16*)alloc(MC * 2);
    float*  kkb   = (float*)alloc(MC * 4);
    float*  vvb   = (float*)alloc(MC * 4);
    float*  rrb   = (float*)alloc(MC * 4);
    float*  wkv   = (float*)alloc(MC * 4);
    __bf16* wkbf  = (__bf16*)alloc(CC * 2);
    __bf16* wvbf  = (__bf16*)alloc(CC * 2);
    __bf16* wrbf  = (__bf16*)alloc(CC * 2);
    __bf16* wobf  = (__bf16*)alloc(CC * 2);
    __bf16* rwkvb = xk;   // reuse: xk dead after the k-GEMM

    {
        int blk = (int)((CC + 255) / 256);
        cvt_f32_bf16<<<blk, 256, 0, stream>>>(kw, wkbf, CC);
        cvt_f32_bf16<<<blk, 256, 0, stream>>>(vw, wvbf, CC);
        cvt_f32_bf16<<<blk, 256, 0, stream>>>(rw, wrbf, CC);
        cvt_f32_bf16<<<blk, 256, 0, stream>>>(ow, wobf, CC);
    }
    ln_stats<<<Mrows, 256, 0, stream>>>(x, mu, rstd);
    mix_kernel<<<(unsigned)((MC + 255) / 256), 256, 0, stream>>>(
        x, mu, rstd, g, beta, tmk, tmv, tmr, xx0, xk, xv, xr);

    dim3 ggrid(Mrows / BM, Cdim / BN);
    gemm_bf16<0><<<ggrid, 256, 0, stream>>>(xk, wkbf, nullptr, kkb, Mrows, Cdim, Cdim);
    gemm_bf16<0><<<ggrid, 256, 0, stream>>>(xv, wvbf, nullptr, vvb, Mrows, Cdim, Cdim);
    gemm_bf16<1><<<ggrid, 256, 0, stream>>>(xr, wrbf, nullptr, rrb, Mrows, Cdim, Cdim);

    wkv_scan<<<(Bdim * Cdim) / 256, 256, 0, stream>>>(kkb, vvb, tf, td, aa0, bb0, pp0, wkv);
    rmul_kernel<<<(unsigned)((MC + 255) / 256), 256, 0, stream>>>(rrb, wkv, rwkvb);
    gemm_bf16<2><<<ggrid, 256, 0, stream>>>(rwkvb, wobf, x, out, Mrows, Cdim, Cdim);
}
